// Attention_44143673868291
// MI455X (gfx1250) — matrix operand order repaired and checked by reading, not compiled, not run on hardware
//
#include <hip/hip_runtime.h>
#include <hip/hip_bf16.h>

typedef __attribute__((ext_vector_type(16))) _Float16 v16h;
typedef __attribute__((ext_vector_type(8)))  _Float16 v8h;
typedef __attribute__((ext_vector_type(8)))  float    v8f;
typedef int v4i __attribute__((vector_size(16)));

static constexpr int Bb  = 8;
static constexpr int Nn  = 2048;
static constexpr int DIM = 256;
static constexpr int POS = 128;
static constexpr int Hh  = 8;
static constexpr int QK  = 384;   // DIM + POS
static constexpr int HD  = 32;    // DIM / H
static constexpr float SCALE = 0.17677669529663687f; // 32^-0.5

// ---------------------------------------------------------------------------
// Async global->LDS copy (16B per lane). gfx1250 GLOBAL_LOAD_ASYNC_TO_LDS_B128,
// tracked by ASYNCcnt. Builtin expects (v4i AS(1)*, v4i AS(3)*, imm off, cpol).
// ---------------------------------------------------------------------------
__device__ __forceinline__ void async_copy_b128(const _Float16* g, _Float16* l) {
#if __has_builtin(__builtin_amdgcn_global_load_async_to_lds_b128)
  __builtin_amdgcn_global_load_async_to_lds_b128(
      (v4i __attribute__((address_space(1)))*)(g),
      (v4i __attribute__((address_space(3)))*)(l), 0, 0);
#else
  *(uint4*)l = *(const uint4*)g;
#endif
}

// ---------------------------------------------------------------------------
// Kernel 0: pack fp32 weight [K][Nout] -> transposed f16 [Nout][K]
// (B-fragment loads in this file take column n in lane n, K contiguous per lane)
// ---------------------------------------------------------------------------
__global__ void k0_pack_wT(const float* __restrict__ src, _Float16* __restrict__ dst,
                           int K, int Nout) {
  int idx = blockIdx.x * blockDim.x + threadIdx.x;
  if (idx >= K * Nout) return;
  int o = idx / K;
  int kk = idx - o * K;
  dst[idx] = (_Float16)src[(size_t)kk * Nout + o];
}

// ---------------------------------------------------------------------------
// Kernel 1: LayerNorm + concat(pos) + Q/K/V projections (WMMA f16 -> f32 acc)
//   q stored [B][H][N][HD] f16, PRE-SCALED by 1/sqrt(HD)
//   k stored [B][H][N][HD] f16
//   v stored transposed [B][H][HD][N] f16 (acts as a natural B-fragment later)
// Block = 256 threads (8 waves), one block per (b, 16-token tile).
// ---------------------------------------------------------------------------
__global__ __launch_bounds__(256) void k1_ln_qkv(
    const float* __restrict__ x, const float* __restrict__ pos,
    const float* __restrict__ lng, const float* __restrict__ lnb,
    const _Float16* __restrict__ WqT, const float* __restrict__ bq,
    const _Float16* __restrict__ WkT, const float* __restrict__ bk,
    const _Float16* __restrict__ WvT, const float* __restrict__ bv,
    _Float16* __restrict__ qo, _Float16* __restrict__ ko,
    _Float16* __restrict__ vto) {
  constexpr int STR = 392;                 // padded row stride (halves)
  __shared__ _Float16 qkin[16 * STR];      // 16 tokens x (256 xn | 128 pos)

  const int blk   = blockIdx.x;
  const int b     = blk >> 7;              // / 128 tiles
  const int tile  = blk & 127;
  const int ibase = tile << 4;
  const int tid   = threadIdx.x;
  const int tloc  = tid >> 4;              // token 0..15
  const int part  = tid & 15;              // 16 threads per token

  // ---- LayerNorm over DIM=256 (16 floats per thread, shfl-xor reduce) ----
  const float* xr = x + ((size_t)(b * Nn + ibase + tloc)) * DIM + part * 16;
  float vals[16];
  float s = 0.f, s2 = 0.f;
#pragma unroll
  for (int i = 0; i < 16; ++i) { float v = xr[i]; vals[i] = v; s += v; s2 += v * v; }
#pragma unroll
  for (int m = 1; m < 16; m <<= 1) { s += __shfl_xor(s, m, 32); s2 += __shfl_xor(s2, m, 32); }
  const float mu   = s * (1.0f / DIM);
  const float var  = s2 * (1.0f / DIM) - mu * mu;
  const float rstd = rsqrtf(var + 1e-5f);
#pragma unroll
  for (int i = 0; i < 16; ++i) {
    int d = part * 16 + i;
    float xn = (vals[i] - mu) * rstd * lng[d] + lnb[d];
    qkin[tloc * STR + d] = (_Float16)xn;
  }
  const float* pr = pos + ((size_t)(b * Nn + ibase + tloc)) * POS + part * 8;
#pragma unroll
  for (int i = 0; i < 8; ++i)
    qkin[tloc * STR + DIM + part * 8 + i] = (_Float16)pr[i];
  __syncthreads();

  const int wave = tid >> 5;
  const int lane = tid & 31;
  const int ln   = lane & 15;
  const int hi   = lane >> 4;

  // 48 output 16-col tiles total (16 q, 16 k, 16 v); 6 per wave
  for (int tt = 0; tt < 6; ++tt) {
    const int t     = wave * 6 + tt;       // 0..47
    const int which = t >> 4;              // 0=q 1=k 2=v
    const int nbase = (t & 15) << 4;
    const _Float16* Wt  = (which == 0) ? WqT : (which == 1) ? WkT : WvT;
    const float*    bia = (which == 0) ? bq  : (which == 1) ? bk  : bv;
    const int Kdim = (which == 2) ? DIM : QK;
    const int nkc  = Kdim >> 5;

    v8f acc = {};
    for (int kc = 0; kc < nkc; ++kc) {
      v16h a, bm;
      const int abase = ln * STR + kc * 32 + hi * 16;
#pragma unroll
      for (int e = 0; e < 8; ++e) { a[e] = qkin[abase + e]; a[8 + e] = qkin[abase + 8 + e]; }
      const _Float16* wr = Wt + (size_t)(nbase + ln) * Kdim + kc * 32 + hi * 16;
#pragma unroll
      for (int e = 0; e < 16; ++e) bm[e] = wr[e];
      acc = __builtin_amdgcn_wmma_f32_16x16x32_f16(false, a, false, bm,
                                                   (short)0, acc, false, false);
    }
    const int c = nbase + ln;
    const float bi = bia[c];
    const int h = c >> 5, d = c & 31;
    if (which == 2) {
      _Float16* dst = vto + ((size_t)((b * Hh + h) * HD + d)) * Nn + ibase + 8 * hi;
#pragma unroll
      for (int r = 0; r < 8; ++r) dst[r] = (_Float16)(acc[r] + bi);
    } else if (which == 0) {
      _Float16* dst = qo + ((size_t)((b * Hh + h) * Nn + ibase + 8 * hi)) * HD + d;
#pragma unroll
      for (int r = 0; r < 8; ++r) dst[(size_t)r * HD] = (_Float16)((acc[r] + bi) * SCALE);
    } else {
      _Float16* dst = ko + ((size_t)((b * Hh + h) * Nn + ibase + 8 * hi)) * HD + d;
#pragma unroll
      for (int r = 0; r < 8; ++r) dst[(size_t)r * HD] = (_Float16)(acc[r] + bi);
    }
  }
}

// ---------------------------------------------------------------------------
// Kernel 2: flash attention, block = 4 waves on the SAME (b,h), 4 adjacent
// 16-query tiles. K / V^T chunks (32 keys) are staged cooperatively into
// double-buffered LDS with GLOBAL_LOAD_ASYNC_TO_LDS_B128 (ASYNCcnt-pipelined),
// then each wave runs 2 score WMMAs + online softmax (shfl-xor row reductions)
// + C->A repack of P through padded LDS + 2 accumulation WMMAs.
// ---------------------------------------------------------------------------
__global__ __launch_bounds__(128) void k2_attn(
    const _Float16* __restrict__ q, const _Float16* __restrict__ k,
    const _Float16* __restrict__ vt, _Float16* __restrict__ O) {
  constexpr int RSTR = 40;                    // padded chunk row stride (halves)
  __shared__ _Float16 kbuf[2][32 * RSTR];     // K chunk:  [key 0..31][d 0..31]
  __shared__ _Float16 vbuf[2][32 * RSTR];     // Vt chunk: [d 0..31][key 0..31]
  __shared__ _Float16 plds_all[4][16 * RSTR]; // per-wave P repack tile

  const int tid  = threadIdx.x;
  const int wave = tid >> 5;
  const int lane = tid & 31;
  const int ln = lane & 15, hi = lane >> 4;
  _Float16* plds = plds_all[wave];

  const int blk = blockIdx.x;                 // B*H*32 = 2048 blocks
  const int b   = blk >> 8;                   // / (H*32)
  const int rem = blk & 255;
  const int h   = rem >> 5;
  const int qt4 = rem & 31;
  const int ibase = ((qt4 << 2) + wave) << 4; // this wave's 16-query tile
  const size_t bh = (size_t)(b * Hh + h);

  const _Float16* kbase = k  + bh * Nn * HD;  // [N][HD]
  const _Float16* vbase = vt + bh * HD * Nn;  // [HD][N]

  // cooperative async staging: 128 threads x 16B per buffer per chunk
  const int row = tid >> 2;                   // 0..31
  const int seg = tid & 3;                    // 16B segment within 64B row

  auto prefetch = [&](int buf, int j0) {
    const _Float16* gk = kbase + (size_t)(j0 + row) * HD + seg * 8;
    const _Float16* gv = vbase + (size_t)row * Nn + j0 + seg * 8;
    async_copy_b128(gk, &kbuf[buf][row * RSTR + seg * 8]);
    async_copy_b128(gv, &vbuf[buf][row * RSTR + seg * 8]);
  };

  // Q A-fragment (pre-scaled by 1/sqrt(HD) in kernel 1)
  v16h qa;
  {
    const _Float16* qp = q + (bh * Nn + ibase + ln) * HD;
    const int k0 = hi * 16;
#pragma unroll
    for (int e = 0; e < 8; ++e) { qa[e] = qp[k0 + e]; qa[8 + e] = qp[8 + k0 + e]; }
  }

  float m_r[8], l_r[8];
  v8f o0 = {}, o1 = {};
#pragma unroll
  for (int r = 0; r < 8; ++r) { m_r[r] = -3.0e38f; l_r[r] = 0.f; }

  prefetch(0, 0);
  for (int jc = 0; jc < Nn / 32; ++jc) {
    const int cur = jc & 1;
    if (jc + 1 < Nn / 32) {
      prefetch(1 - cur, (jc + 1) * 32);
      asm volatile("s_wait_asynccnt 2" ::: "memory"); // chunk jc landed; next in flight
    } else {
      asm volatile("s_wait_asynccnt 0" ::: "memory");
    }
    __syncthreads();                           // buffer `cur` visible block-wide

    // K B-fragments (col n=ln keys, K-dim = head dim, contiguous per lane)
    v16h kb0, kb1, vb0, vb1;
    {
      const _Float16* kr0 = &kbuf[cur][ln * RSTR + hi * 16];
      const _Float16* kr1 = &kbuf[cur][(16 + ln) * RSTR + hi * 16];
      const _Float16* vr0 = &vbuf[cur][ln * RSTR + hi * 16];
      const _Float16* vr1 = &vbuf[cur][(16 + ln) * RSTR + hi * 16];
#pragma unroll
      for (int e = 0; e < 16; ++e) {
        kb0[e] = kr0[e]; kb1[e] = kr1[e];
        vb0[e] = vr0[e]; vb1[e] = vr1[e];
      }
    }
    const v8f z = {};
    v8f s0 = __builtin_amdgcn_wmma_f32_16x16x32_f16(false, qa, false, kb0, (short)0, z, false, false);
    v8f s1 = __builtin_amdgcn_wmma_f32_16x16x32_f16(false, qa, false, kb1, (short)0, z, false, false);

    // ---- online softmax (row = r + 8*hi, col = ln in C layout) ----
#pragma unroll
    for (int r = 0; r < 8; ++r) {
      float mx = fmaxf(s0[r], s1[r]);
#pragma unroll
      for (int mm = 1; mm < 16; mm <<= 1) mx = fmaxf(mx, __shfl_xor(mx, mm, 32));
      const float mnew = fmaxf(m_r[r], mx);
      const float corr = __expf(m_r[r] - mnew);
      m_r[r] = mnew;
      const float p0 = __expf(s0[r] - mnew);
      const float p1 = __expf(s1[r] - mnew);
      float rs = p0 + p1;
#pragma unroll
      for (int mm = 1; mm < 16; mm <<= 1) rs += __shfl_xor(rs, mm, 32);
      l_r[r] = l_r[r] * corr + rs;
      o0[r] *= corr;
      o1[r] *= corr;
      const int m = r + 8 * hi;
      plds[m * RSTR + ln]      = (_Float16)p0;
      plds[m * RSTR + 16 + ln] = (_Float16)p1;
    }
    asm volatile("s_wait_dscnt 0" ::: "memory");  // wave-local LDS RAW fence

    // P as A-fragment from LDS
    v16h pa;
    {
      const int base = ln * RSTR + hi * 16;
#pragma unroll
      for (int e = 0; e < 8; ++e) { pa[e] = plds[base + e]; pa[8 + e] = plds[base + 8 + e]; }
    }
    o0 = __builtin_amdgcn_wmma_f32_16x16x32_f16(false, pa, false, vb0, (short)0, o0, false, false);
    o1 = __builtin_amdgcn_wmma_f32_16x16x32_f16(false, pa, false, vb1, (short)0, o1, false, false);

    __syncthreads();  // everyone done reading `cur` before it is re-filled
  }

  // finalize: divide by l, store O [B][N][DIM] f16 (A-friendly for kernel 3)
  _Float16* op = O + (size_t)b * Nn * DIM;
#pragma unroll
  for (int r = 0; r < 8; ++r) {
    const float inv = 1.0f / l_r[r];
    const int i = ibase + r + 8 * hi;
    op[(size_t)i * DIM + h * HD + ln]      = (_Float16)(o0[r] * inv);
    op[(size_t)i * DIM + h * HD + 16 + ln] = (_Float16)(o1[r] * inv);
  }
}

// ---------------------------------------------------------------------------
// Kernel 3: out = x + O @ Wp + bp   (WMMA, A straight from global f16 O)
// Block = 256 (8 waves), one block per (b, 16-token tile); 2 col-tiles/wave.
// ---------------------------------------------------------------------------
__global__ __launch_bounds__(256) void k3_proj(
    const _Float16* __restrict__ O, const _Float16* __restrict__ WpT,
    const float* __restrict__ bp, const float* __restrict__ x,
    float* __restrict__ out) {
  const int blk   = blockIdx.x;
  const int b     = blk >> 7;
  const int tile  = blk & 127;
  const int ibase = tile << 4;
  const int wave  = threadIdx.x >> 5;
  const int lane  = threadIdx.x & 31;
  const int ln = lane & 15, hi = lane >> 4;

  const _Float16* orow = O + ((size_t)(b * Nn + ibase + ln)) * DIM;

  for (int tt = 0; tt < 2; ++tt) {
    const int nbase = (wave * 2 + tt) << 4;
    v8f acc = {};
    for (int kc = 0; kc < 8; ++kc) {
      v16h a, bm;
      const int k0 = kc * 32 + hi * 16;
#pragma unroll
      for (int e = 0; e < 8; ++e) { a[e] = orow[k0 + e]; a[8 + e] = orow[k0 + 8 + e]; }
      const _Float16* wr = WpT + (size_t)(nbase + ln) * DIM + kc * 32 + hi * 16;
#pragma unroll
      for (int e = 0; e < 16; ++e) bm[e] = wr[e];
      acc = __builtin_amdgcn_wmma_f32_16x16x32_f16(false, a, false, bm,
                                                   (short)0, acc, false, false);
    }
    const int c = nbase + ln;
    const float bi = bp[c];
#pragma unroll
    for (int r = 0; r < 8; ++r) {
      const int i = ibase + r + 8 * hi;
      const size_t idx = ((size_t)(b * Nn + i)) * DIM + c;
      out[idx] = acc[r] + bi + x[idx];
    }
  }
}

// ---------------------------------------------------------------------------
extern "C" void kernel_launch(void* const* d_in, const int* in_sizes, int n_in,
                              void* d_out, int out_size, void* d_ws, size_t ws_size,
                              hipStream_t stream) {
  const float* x    = (const float*)d_in[0];
  const float* pos  = (const float*)d_in[1];
  const float* lng  = (const float*)d_in[2];
  const float* lnb  = (const float*)d_in[3];
  const float* Wq   = (const float*)d_in[4];
  const float* bq   = (const float*)d_in[5];
  const float* Wk   = (const float*)d_in[6];
  const float* bk   = (const float*)d_in[7];
  const float* Wv   = (const float*)d_in[8];
  const float* bv   = (const float*)d_in[9];
  const float* Wp   = (const float*)d_in[10];
  const float* bp   = (const float*)d_in[11];

  char* ws = (char*)d_ws;
  constexpr size_t szWqkT = (size_t)DIM * QK * 2;          // 192 KiB
  constexpr size_t szWsqT = (size_t)DIM * DIM * 2;         // 128 KiB
  constexpr size_t szAct  = (size_t)Bb * Hh * Nn * HD * 2; // 8 MiB each
  _Float16* WqT = (_Float16*)(ws);
  _Float16* WkT = (_Float16*)(ws + szWqkT);
  _Float16* WvT = (_Float16*)(ws + 2 * szWqkT);
  _Float16* WpT = (_Float16*)(ws + 2 * szWqkT + szWsqT);
  _Float16* qo  = (_Float16*)(ws + 2 * szWqkT + 2 * szWsqT);
  _Float16* ko  = (_Float16*)(ws + 2 * szWqkT + 2 * szWsqT + 1 * szAct);
  _Float16* vto = (_Float16*)(ws + 2 * szWqkT + 2 * szWsqT + 2 * szAct);
  _Float16* Ob  = (_Float16*)(ws + 2 * szWqkT + 2 * szWsqT + 3 * szAct);

  k0_pack_wT<<<(QK * DIM + 255) / 256, 256, 0, stream>>>(Wq, WqT, QK, DIM);
  k0_pack_wT<<<(QK * DIM + 255) / 256, 256, 0, stream>>>(Wk, WkT, QK, DIM);
  k0_pack_wT<<<(DIM * DIM + 255) / 256, 256, 0, stream>>>(Wv, WvT, DIM, DIM);
  k0_pack_wT<<<(DIM * DIM + 255) / 256, 256, 0, stream>>>(Wp, WpT, DIM, DIM);

  k1_ln_qkv<<<Bb * (Nn / 16), 256, 0, stream>>>(x, pos, lng, lnb,
                                                WqT, bq, WkT, bk, WvT, bv,
                                                qo, ko, vto);
  k2_attn<<<Bb * Hh * (Nn / 64), 128, 0, stream>>>(qo, ko, vto, Ob);
  k3_proj<<<Bb * (Nn / 16), 256, 0, stream>>>(Ob, WpT, bp, x, (float*)d_out);
}
